// SimpleMambaLike_81595788689874
// MI455X (gfx1250) — matrix operand order repaired and checked by reading, not compiled, not run on hardware
//
#include <hip/hip_runtime.h>
#include <hip/hip_bf16.h>
#include <stdint.h>

// ---------------- problem constants ----------------
#define D_MODEL 1024
#define D_CONV  4
#define D_INNER 2048
#define BATCH   4
#define SEQ     2048
#define ROWS    (BATCH * SEQ)        // 8192 flattened (b, t) rows
#define CHUNK   128                  // scan chunk length
#define NCHUNK  (SEQ / CHUNK)        // 16 chunks per sequence

typedef __attribute__((ext_vector_type(16))) __bf16 v16bf;
typedef __attribute__((ext_vector_type(8)))  float  v8f;

// ---------------- bf16 helpers (bit-exact RNE) ----------------
__device__ __forceinline__ unsigned short f2bf(float f) {
    unsigned u = __float_as_uint(f);
    unsigned r = 0x7FFFu + ((u >> 16) & 1u);
    return (unsigned short)((u + r) >> 16);
}
__device__ __forceinline__ float bf2f(unsigned short h) {
    return __uint_as_float(((unsigned)h) << 16);
}
// Two adjacent bf16 values in one 32-bit load.
__device__ __forceinline__ float2 bfpair(const unsigned short* __restrict__ p) {
    unsigned u = *(const unsigned*)p;
    return make_float2(__uint_as_float(u << 16), __uint_as_float(u & 0xFFFF0000u));
}
__device__ __forceinline__ unsigned packbf(float lo, float hi) {
    return (unsigned)f2bf(lo) | ((unsigned)f2bf(hi) << 16);
}

union FragU { v16bf v; uint4 q[2]; };

// A fragment: 16x32 bf16, A row-major (lda = K).
// Lanes 0-15: M=lane, elems 0..7 = K[k0..k0+7], elems 8..15 = K[k0+16..k0+23]
// Lanes 16-31: M=lane-16, K runs shifted by +8.
__device__ __forceinline__ v16bf load_a_frag(const unsigned short* __restrict__ A,
                                             int lda, int row0, int k0, int lane) {
    int r  = row0 + (lane & 15);
    int ks = k0 + ((lane & 16) ? 8 : 0);
    const unsigned short* p = A + (size_t)r * lda + ks;
    FragU u;
    u.q[0] = *(const uint4*)(p);
    u.q[1] = *(const uint4*)(p + 16);
    return u.v;
}

// B fragment: 32x16 bf16 from N-major (transposed) weights Bt[n][k], ldb = K.
// Lanes 0-15: N=lane, elems 0..7 = K[k0..k0+7], elems 8..15 = K[k0+16..k0+23]; lanes 16-31: N=lane-16, K runs shifted by +8.
__device__ __forceinline__ v16bf load_b_frag(const unsigned short* __restrict__ Bt,
                                             int ldb, int n0, int k0, int lane) {
    int n  = n0 + (lane & 15);
    int ks = k0 + ((lane & 16) ? 8 : 0);
    const unsigned short* p = Bt + (size_t)n * ldb + ks;
    FragU u;
    u.q[0] = *(const uint4*)(p);
    u.q[1] = *(const uint4*)(p + 16);
    return u.v;
}

// ---------------- GEMM: C[M,N] = A[M,K](bf16) * Bt[N,K](bf16)^T + bias ----------------
// Workgroup: 512 threads = 16 waves (4M x 4N); WG tile 128(M) x 256(N);
// wave tile 32x64 (2x4 WMMA accumulators).
__global__ __launch_bounds__(512)
void gemm_bf16_kernel(const unsigned short* __restrict__ A,
                      const unsigned short* __restrict__ Bt,
                      const float* __restrict__ bias,
                      float* __restrict__ Cf,            // fp32 output (or null)
                      unsigned short* __restrict__ Cb,   // bf16 output (or null)
                      int M, int N, int K) {
    const int lane = threadIdx.x & 31;
    const int wave = threadIdx.x >> 5;
    const int wm = wave & 3;      // 0..3 -> M offset
    const int wn = wave >> 2;     // 0..3 -> N offset
    const int mw = blockIdx.y * 128 + wm * 32;
    const int nw = blockIdx.x * 256 + wn * 64;

    v8f acc[2][4];
    const v8f zf = {0.f, 0.f, 0.f, 0.f, 0.f, 0.f, 0.f, 0.f};
#pragma unroll
    for (int i = 0; i < 2; ++i)
#pragma unroll
        for (int j = 0; j < 4; ++j) acc[i][j] = zf;

    for (int k0 = 0; k0 < K; k0 += 32) {
        v16bf a0 = load_a_frag(A, K, mw,      k0, lane);
        v16bf a1 = load_a_frag(A, K, mw + 16, k0, lane);
        v16bf b0 = load_b_frag(Bt, K, nw,      k0, lane);
        v16bf b1 = load_b_frag(Bt, K, nw + 16, k0, lane);
        v16bf b2 = load_b_frag(Bt, K, nw + 32, k0, lane);
        v16bf b3 = load_b_frag(Bt, K, nw + 48, k0, lane);

        acc[0][0] = __builtin_amdgcn_wmma_f32_16x16x32_bf16(false, a0, false, b0, (short)0, acc[0][0], false, false);
        acc[0][1] = __builtin_amdgcn_wmma_f32_16x16x32_bf16(false, a0, false, b1, (short)0, acc[0][1], false, false);
        acc[0][2] = __builtin_amdgcn_wmma_f32_16x16x32_bf16(false, a0, false, b2, (short)0, acc[0][2], false, false);
        acc[0][3] = __builtin_amdgcn_wmma_f32_16x16x32_bf16(false, a0, false, b3, (short)0, acc[0][3], false, false);
        acc[1][0] = __builtin_amdgcn_wmma_f32_16x16x32_bf16(false, a1, false, b0, (short)0, acc[1][0], false, false);
        acc[1][1] = __builtin_amdgcn_wmma_f32_16x16x32_bf16(false, a1, false, b1, (short)0, acc[1][1], false, false);
        acc[1][2] = __builtin_amdgcn_wmma_f32_16x16x32_bf16(false, a1, false, b2, (short)0, acc[1][2], false, false);
        acc[1][3] = __builtin_amdgcn_wmma_f32_16x16x32_bf16(false, a1, false, b3, (short)0, acc[1][3], false, false);
    }

    // C/D layout: VGPR r holds M = r (lanes 0-15) / r+8 (lanes 16-31); N = lane%16.
    const int roff = (lane & 16) ? 8 : 0;
    const int col  = lane & 15;
#pragma unroll
    for (int i = 0; i < 2; ++i) {
#pragma unroll
        for (int j = 0; j < 4; ++j) {
            int ncol = nw + j * 16 + col;
            float bv = bias[ncol];
#pragma unroll
            for (int r = 0; r < 8; ++r) {
                int row = mw + i * 16 + roff + r;
                float v = acc[i][j][r] + bv;
                size_t idx = (size_t)row * N + ncol;
                if (Cb) Cb[idx] = f2bf(v);
                else    Cf[idx] = v;
            }
        }
    }
}

// ---------------- fp32 -> bf16 convert, 2 elements/thread ----------------
__global__ void cvt_f32_bf16_kernel(const float* __restrict__ in,
                                    unsigned* __restrict__ out, size_t npair) {
    size_t i = (size_t)blockIdx.x * blockDim.x + threadIdx.x;
    if (i < npair) {
        float2 v = *(const float2*)(in + 2 * i);
        out[i] = packbf(v.x, v.y);
    }
}

// ---------------- W[K,N] fp32 -> Wt[N,K] bf16 (LDS tiled transpose) ----------------
__global__ void transpose_cvt_kernel(const float* __restrict__ W,
                                     unsigned short* __restrict__ Wt, int K, int N) {
    __shared__ float tile[32][33];
    int kb = blockIdx.y * 32, nb = blockIdx.x * 32;
    int tx = threadIdx.x;  // 0..31
    int ty = threadIdx.y;  // 0..7
    for (int i = ty; i < 32; i += 8)
        tile[i][tx] = W[(size_t)(kb + i) * N + nb + tx];
    __syncthreads();
    for (int i = ty; i < 32; i += 8)
        Wt[(size_t)(nb + i) * K + kb + tx] = f2bf(tile[tx][i]);
}

// ---------------- depthwise causal conv (k=4) + bias + SiLU, 2 channels/thread -------
// ug: [ROWS, 2*D_INNER] bf16, u = cols [0, D_INNER). Output uconv: [ROWS, D_INNER] bf16.
__global__ void conv_silu_kernel(const unsigned short* __restrict__ ug,
                                 const float* __restrict__ conv_w,
                                 const float* __restrict__ conv_b,
                                 unsigned* __restrict__ uconv) {
    size_t idx = (size_t)blockIdx.x * blockDim.x + threadIdx.x;   // pair index
    const size_t total = (size_t)ROWS * (D_INNER / 2);
    if (idx >= total) return;
    int chp       = (int)(idx % (D_INNER / 2));   // channel pair
    int ch        = chp * 2;
    size_t rowidx = idx / (D_INNER / 2);          // bat*SEQ + t
    int t         = (int)(rowidx % SEQ);

    float4 wa = *(const float4*)(conv_w + ch * 4);        // ch   taps 0..3
    float4 wb = *(const float4*)(conv_w + (ch + 1) * 4);  // ch+1 taps 0..3
    float2 cb = *(const float2*)(conv_b + ch);

    const size_t ld = 2 * D_INNER;
    float s0 = cb.x, s1 = cb.y;
    if (t >= 3) { float2 u = bfpair(ug + (rowidx - 3) * ld + ch); s0 += wa.x * u.x; s1 += wb.x * u.y; }
    if (t >= 2) { float2 u = bfpair(ug + (rowidx - 2) * ld + ch); s0 += wa.y * u.x; s1 += wb.y * u.y; }
    if (t >= 1) { float2 u = bfpair(ug + (rowidx - 1) * ld + ch); s0 += wa.z * u.x; s1 += wb.z * u.y; }
    { float2 u = bfpair(ug + rowidx * ld + ch); s0 += wa.w * u.x; s1 += wb.w * u.y; }

    float y0 = s0 / (1.f + __expf(-s0));   // SiLU
    float y1 = s1 / (1.f + __expf(-s1));
    uconv[rowidx * (D_INNER / 2) + chp] = packbf(y0, y1);
}

// ---------------- chunked SSM scan ----------------
// a = exp(-softplus(ld)) = 1/(1+exp(ld)) = sigmoid(-ld)
__device__ __forceinline__ float decay_a(float ld) {
    return 1.f / (1.f + __expf(ld));
}

// Pass 1: per-chunk local scan (zero init), emit final local state (carry).
// 2 channels per thread. Threads: BATCH * NCHUNK * D_INNER/2.
__global__ __launch_bounds__(256)
void scan_pass1_kernel(const unsigned short* __restrict__ bc,
                       const float* __restrict__ log_decay,
                       float2* __restrict__ carry) {
    int tid = blockIdx.x * blockDim.x + threadIdx.x;
    if (tid >= BATCH * NCHUNK * (D_INNER / 2)) return;
    int chp = tid % (D_INNER / 2);
    int ch  = chp * 2;
    int j   = (tid / (D_INNER / 2)) % NCHUNK;
    int bat = tid / ((D_INNER / 2) * NCHUNK);

    float2 ld = *(const float2*)(log_decay + ch);
    float a0 = decay_a(ld.x), a1 = decay_a(ld.y);
    float s0 = 0.f, s1 = 0.f;
    const size_t ld2 = 2 * D_INNER;
    size_t row = (size_t)bat * SEQ + (size_t)j * CHUNK;
    for (int t = 0; t < CHUNK; ++t, ++row) {
        float2 bb = bfpair(bc + row * ld2 + ch);
        s0 = a0 * s0 + bb.x;
        s1 = a1 * s1 + bb.y;
    }
    carry[((size_t)bat * NCHUNK + j) * (D_INNER / 2) + chp] = make_float2(s0, s1);
}

// Pass 2: serial prefix over the 16 chunk carries per (bat, ch pair).
__global__ __launch_bounds__(256)
void scan_pass2_kernel(const float2* __restrict__ carry,
                       const float* __restrict__ log_decay,
                       float2* __restrict__ incoming) {
    int tid = blockIdx.x * blockDim.x + threadIdx.x;
    if (tid >= BATCH * (D_INNER / 2)) return;
    int chp = tid % (D_INNER / 2);
    int ch  = chp * 2;
    int bat = tid / (D_INNER / 2);

    float2 ld = *(const float2*)(log_decay + ch);
    float a0 = decay_a(ld.x), a1 = decay_a(ld.y);
    float aL0 = __powf(a0, (float)CHUNK);
    float aL1 = __powf(a1, (float)CHUNK);
    float i0 = 0.f, i1 = 0.f;
#pragma unroll
    for (int j = 0; j < NCHUNK; ++j) {
        size_t idx = ((size_t)bat * NCHUNK + j) * (D_INNER / 2) + chp;
        incoming[idx] = make_float2(i0, i1);
        float2 c = carry[idx];
        i0 = aL0 * i0 + c.x;
        i1 = aL1 * i1 + c.y;
    }
}

// Pass 3: re-scan each chunk seeded with exact incoming state; y = state*c,
// gate with sigmoid(gate), write bf16 pairs.
__global__ __launch_bounds__(256)
void scan_pass3_kernel(const unsigned short* __restrict__ bc,
                       const unsigned short* __restrict__ ug,
                       const float* __restrict__ log_decay,
                       const float2* __restrict__ incoming,
                       unsigned* __restrict__ yg) {
    int tid = blockIdx.x * blockDim.x + threadIdx.x;
    if (tid >= BATCH * NCHUNK * (D_INNER / 2)) return;
    int chp = tid % (D_INNER / 2);
    int ch  = chp * 2;
    int j   = (tid / (D_INNER / 2)) % NCHUNK;
    int bat = tid / ((D_INNER / 2) * NCHUNK);

    float2 ld = *(const float2*)(log_decay + ch);
    float a0 = decay_a(ld.x), a1 = decay_a(ld.y);
    float2 inc = incoming[((size_t)bat * NCHUNK + j) * (D_INNER / 2) + chp];
    float s0 = inc.x, s1 = inc.y;
    const size_t ld2 = 2 * D_INNER;
    size_t row = (size_t)bat * SEQ + (size_t)j * CHUNK;
    for (int t = 0; t < CHUNK; ++t, ++row) {
        float2 bb = bfpair(bc + row * ld2 + ch);
        float2 cc = bfpair(bc + row * ld2 + D_INNER + ch);
        float2 gg = bfpair(ug + row * ld2 + D_INNER + ch);
        s0 = a0 * s0 + bb.x;
        s1 = a1 * s1 + bb.y;
        float y0 = s0 * cc.x * (1.f / (1.f + __expf(-gg.x)));
        float y1 = s1 * cc.y * (1.f / (1.f + __expf(-gg.y)));
        yg[row * (D_INNER / 2) + chp] = packbf(y0, y1);
    }
}

// ---------------- launch ----------------
extern "C" void kernel_launch(void* const* d_in, const int* in_sizes, int n_in,
                              void* d_out, int out_size, void* d_ws, size_t ws_size,
                              hipStream_t stream) {
    const float* x         = (const float*)d_in[0];
    const float* W_in      = (const float*)d_in[1];
    const float* b_in      = (const float*)d_in[2];
    const float* conv_w    = (const float*)d_in[3];
    const float* conv_b    = (const float*)d_in[4];
    const float* W_bc      = (const float*)d_in[5];
    const float* b_bc      = (const float*)d_in[6];
    const float* log_decay = (const float*)d_in[7];
    const float* W_out     = (const float*)d_in[8];
    const float* b_out     = (const float*)d_in[9];
    (void)in_sizes; (void)n_in; (void)out_size; (void)ws_size;

    char* ws = (char*)d_ws;
    size_t off = 0;
    auto alloc = [&](size_t bytes) -> char* {
        char* p = ws + off;
        off += (bytes + 255) & ~(size_t)255;
        return p;
    };
    unsigned short* xb     = (unsigned short*)alloc((size_t)ROWS * D_MODEL * 2);          // 16 MB
    unsigned short* Wt_in  = (unsigned short*)alloc((size_t)(2 * D_INNER) * D_MODEL * 2); // 8 MB
    unsigned short* Wt_bc  = (unsigned short*)alloc((size_t)(2 * D_INNER) * D_INNER * 2); // 16 MB
    unsigned short* Wt_out = (unsigned short*)alloc((size_t)D_MODEL * D_INNER * 2);       // 4 MB
    unsigned short* ug     = (unsigned short*)alloc((size_t)ROWS * 2 * D_INNER * 2);      // 64 MB
    unsigned short* uconv  = (unsigned short*)alloc((size_t)ROWS * D_INNER * 2);          // 32 MB
    unsigned short* bcbuf  = (unsigned short*)alloc((size_t)ROWS * 2 * D_INNER * 2);      // 64 MB
    unsigned short* yg     = (unsigned short*)alloc((size_t)ROWS * D_INNER * 2);          // 32 MB
    float2*         carry  = (float2*)alloc((size_t)BATCH * NCHUNK * (D_INNER / 2) * 8);  // 512 KB
    float2*         incb   = (float2*)alloc((size_t)BATCH * NCHUNK * (D_INNER / 2) * 8);  // 512 KB

    // 1) precision conversion / weight transposition
    {
        size_t npair = (size_t)ROWS * D_MODEL / 2;
        cvt_f32_bf16_kernel<<<(unsigned)((npair + 255) / 256), 256, 0, stream>>>(
            x, (unsigned*)xb, npair);
    }
    dim3 tb(32, 8);
    transpose_cvt_kernel<<<dim3((2 * D_INNER) / 32, D_MODEL / 32), tb, 0, stream>>>(W_in,  Wt_in,  D_MODEL, 2 * D_INNER);
    transpose_cvt_kernel<<<dim3((2 * D_INNER) / 32, D_INNER / 32), tb, 0, stream>>>(W_bc,  Wt_bc,  D_INNER, 2 * D_INNER);
    transpose_cvt_kernel<<<dim3(D_MODEL / 32,       D_INNER / 32), tb, 0, stream>>>(W_out, Wt_out, D_INNER, D_MODEL);

    // 2) ug = x @ W_in + b_in   (bf16 out)
    gemm_bf16_kernel<<<dim3((2 * D_INNER) / 256, ROWS / 128), 512, 0, stream>>>(
        xb, Wt_in, b_in, nullptr, ug, ROWS, 2 * D_INNER, D_MODEL);

    // 3) depthwise causal conv + SiLU
    {
        size_t npair = (size_t)ROWS * D_INNER / 2;
        conv_silu_kernel<<<(unsigned)((npair + 255) / 256), 256, 0, stream>>>(
            ug, conv_w, conv_b, (unsigned*)uconv);
    }

    // 4) bc = u_conv @ W_bc + b_bc   (bf16 out)
    gemm_bf16_kernel<<<dim3((2 * D_INNER) / 256, ROWS / 128), 512, 0, stream>>>(
        uconv, Wt_bc, b_bc, nullptr, bcbuf, ROWS, 2 * D_INNER, D_INNER);

    // 5) chunked SSM scan + gating (exact two-level recombination)
    {
        int n1 = BATCH * NCHUNK * (D_INNER / 2);   // 65536 threads
        scan_pass1_kernel<<<(n1 + 255) / 256, 256, 0, stream>>>(bcbuf, log_decay, carry);
        int n2 = BATCH * (D_INNER / 2);            // 4096 threads
        scan_pass2_kernel<<<(n2 + 255) / 256, 256, 0, stream>>>(carry, log_decay, incb);
        scan_pass3_kernel<<<(n1 + 255) / 256, 256, 0, stream>>>(bcbuf, ug, log_decay, incb, (unsigned*)yg);
    }

    // 6) out = y_gated @ W_out + b_out   (fp32 out -> d_out)
    gemm_bf16_kernel<<<dim3(D_MODEL / 256, ROWS / 128), 512, 0, stream>>>(
        yg, Wt_out, b_out, (float*)d_out, nullptr, ROWS, D_MODEL, D_INNER);
}
